// MultiHeadAttention_21809843929563
// MI455X (gfx1250) — matrix operand order repaired and checked by reading, not compiled, not run on hardware
//
#include <hip/hip_runtime.h>
#include <hip/hip_bf16.h>

typedef __attribute__((ext_vector_type(8)))  __bf16 v8bf;
typedef __attribute__((ext_vector_type(16))) __bf16 v16bf;
typedef __attribute__((ext_vector_type(8)))  float  v8f;
typedef __attribute__((ext_vector_type(4)))  int    v4i;

#define EMB    768
#define SEQ    4096
#define BATCH  2
#define MTOT   (BATCH*SEQ)      // 8192 rows (b*n flattened)
#define HEADS  8
#define HDIM   96
#define KSTEPS (EMB/32)         // 24 k-steps of 32
#define SCALE_F 0.1020620726159658f   // 96^-0.5 (applied AFTER softmax, per reference)

#if defined(__has_builtin)
#if __has_builtin(__builtin_amdgcn_global_load_async_to_lds_b128)
#define HAVE_ASYNC_LDS 1
#endif
#endif
#ifndef HAVE_ASYNC_LDS
#define HAVE_ASYNC_LDS 0
#endif

typedef __attribute__((address_space(1))) v4i gv4i;
typedef __attribute__((address_space(3))) v4i lv4i;

__device__ __forceinline__ v16bf join16(v8bf lo, v8bf hi) {
  union { v16bf v; v8bf h[2]; } u;
  u.h[0] = lo; u.h[1] = hi;
  return u.v;
}

__device__ __forceinline__ v8f wmma_bf16(v16bf a, v16bf b, v8f c) {
  return __builtin_amdgcn_wmma_f32_16x16x32_bf16(false, a, false, b, (short)0, c, false, false);
}

// 16-byte global -> LDS copy; async (ASYNCcnt) when the builtin exists.
__device__ __forceinline__ void cp16(const __bf16* g, __bf16* l) {
#if HAVE_ASYNC_LDS
  __builtin_amdgcn_global_load_async_to_lds_b128((gv4i*)g, (lv4i*)l, 0, 0);
#else
  *(v8bf*)l = *(const v8bf*)g;
#endif
}

__device__ __forceinline__ void drain_async() {
#if HAVE_ASYNC_LDS
  asm volatile("s_wait_asynccnt 0x0" ::: "memory");
#endif
}

// ---------------------------------------------------------------- fp32 -> bf16
__global__ void cvt_bf16_kernel(const float* __restrict__ src,
                                __bf16* __restrict__ dst, int n) {
  int i = blockIdx.x * blockDim.x + threadIdx.x;
  int stride = gridDim.x * blockDim.x;
  for (; i < n; i += stride) dst[i] = (__bf16)src[i];
}

// ------------------------------------------- pack W[K=768][N=768] row-major ->
// B-fragment blocks: dst[((nt*KSTEPS + kt)*32 + lane)*16 + i]
__global__ void pack_w_kernel(const float* __restrict__ W, __bf16* __restrict__ dst) {
  int p = blockIdx.x * blockDim.x + threadIdx.x;
  int i    = p & 15;
  int lane = (p >> 4) & 31;
  int kt   = (p >> 9) % KSTEPS;
  int nt   = p / (512 * KSTEPS);
  int col  = nt * 16 + (lane & 15);
  int kk   = kt * 32 + (i >> 3) * 16 + (lane >> 4) * 8 + (i & 7);
  dst[p] = (__bf16)W[kk * EMB + col];
}

// ------------------------------------------------------------- WMMA GEMM
// C[8192x768] = A(bf16 rm) @ Bpacked + bias. 8 waves (4x2), block 128x128,
// wave tile 32x64 (2 row-frags x 4 col-tiles = 8 wmma / k-step).
// MODE 0: bf16 row-major   MODE 1: bf16 per-head-transposed V   MODE 2: fp32
template <int MODE>
__global__ __launch_bounds__(256) void gemm_kernel(
    const __bf16* __restrict__ A, const __bf16* __restrict__ Bp,
    const float* __restrict__ bias, void* __restrict__ out) {
  const int lane  = threadIdx.x & 31;
  const int wave  = threadIdx.x >> 5;
  const int wm    = wave >> 1;
  const int wn    = wave & 1;
  const int rbase = blockIdx.x * 128 + wm * 32;
  const int cbase = blockIdx.y * 128 + wn * 64;
  const int hsel  = lane >> 4;
  const int l15   = lane & 15;

  v8f acc[2][4] = {};
  const __bf16* ap0 = A + (size_t)(rbase + l15) * EMB + hsel * 8;
  const __bf16* ap1 = ap0 + (size_t)16 * EMB;
  const __bf16* bp = Bp + ((size_t)(cbase >> 4) * KSTEPS * 32 + lane) * 16;

  for (int kt = 0; kt < KSTEPS; ++kt) {
    v16bf a0 = join16(*(const v8bf*)(ap0), *(const v8bf*)(ap0 + 16));
    v16bf a1 = join16(*(const v8bf*)(ap1), *(const v8bf*)(ap1 + 16));
    ap0 += 32; ap1 += 32;
#pragma unroll
    for (int t = 0; t < 4; ++t) {
      v16bf b = *(const v16bf*)(bp + t * (KSTEPS * 512));
      acc[0][t] = wmma_bf16(a0, b, acc[0][t]);
      acc[1][t] = wmma_bf16(a1, b, acc[1][t]);
    }
    bp += 512;
  }

#pragma unroll
  for (int t = 0; t < 4; ++t) {
    const int col = cbase + t * 16 + l15;
    const float bv = bias[col];
#pragma unroll
    for (int g = 0; g < 2; ++g) {
#pragma unroll
      for (int r = 0; r < 8; ++r) {
        const int row = rbase + g * 16 + hsel * 8 + r;
        const float vv = acc[g][t][r] + bv;
        if (MODE == 0) {
          ((__bf16*)out)[(size_t)row * EMB + col] = (__bf16)vv;
        } else if (MODE == 1) {
          const int b = row >> 12, n = row & (SEQ - 1);
          const int h = col / HDIM, d = col % HDIM;
          ((__bf16*)out)[((size_t)((b * HEADS + h) * HDIM + d)) * SEQ + n] = (__bf16)vv;
        } else {
          ((float*)out)[(size_t)row * EMB + col] = vv;
        }
      }
    }
  }
}

// ------------------------------------------------------------- flash attention
// Q,K bf16 [8192][768] rm; Vt bf16 [B*H][96][4096]; Out bf16 [8192][768].
// Block = 8 waves sharing one (b,h): K/V 64-key tiles staged once per block
// into LDS via async global->LDS copies; each wave owns 16 query rows.
#define SKP 104   // sK row stride (64 rows x 96 cols, bank-conflict pad)
#define SVP 72    // sV row stride (96 rows x 64 cols)
#define SPP 72    // sP row stride (per-wave 16 rows x 64 cols)
__global__ __launch_bounds__(256) void flash_kernel(
    const __bf16* __restrict__ Q, const __bf16* __restrict__ Kmat,
    const __bf16* __restrict__ Vt, __bf16* __restrict__ Out) {
  __shared__ __bf16 sK[64 * SKP];        // 13.0 KB
  __shared__ __bf16 sV[96 * SVP];        // 13.5 KB
  __shared__ __bf16 sP[8 * 16 * SPP];    // 18.0 KB
  const int tid  = threadIdx.x;
  const int lane = tid & 31;
  const int wave = tid >> 5;
  const int l15  = lane & 15;
  const int hsel = lane >> 4;
  const int bh   = blockIdx.y;
  const int b    = bh >> 3;
  const int h    = bh & 7;
  const int n0   = blockIdx.x * 128 + wave * 16;
  const int gq   = b * SEQ + n0;

  // Q fragments for the 3 k-chunks of head-dim 96
  v16bf aq[3];
  {
    const __bf16* qrow = Q + (size_t)(gq + l15) * EMB + h * HDIM + hsel * 8;
#pragma unroll
    for (int dc = 0; dc < 3; ++dc)
      aq[dc] = join16(*(const v8bf*)(qrow + dc * 32),
                      *(const v8bf*)(qrow + dc * 32 + 16));
  }

  v8f o[6] = {};
  float m[8], l[8];
#pragma unroll
  for (int r = 0; r < 8; ++r) { m[r] = -1e30f; l[r] = 0.f; }

  __bf16* pl = sP + wave * 16 * SPP;
  const __bf16* kg = Kmat + (size_t)(b * SEQ) * EMB + h * HDIM;  // +(kb+row)*EMB+col
  const __bf16* vg = Vt + (size_t)bh * HDIM * SEQ;               // +row*SEQ+kb+col

  for (int kb = 0; kb < SEQ; kb += 64) {
    // ---- cooperative stage of K (64x96) and V (96x64) tiles into LDS
    //      768 16B-chunks each; 3 chunks per thread per tile.
#pragma unroll
    for (int t = 0; t < 3; ++t) {
      const int c  = tid + t * 256;
      const int kr = c / 12, kc = (c % 12) * 8;      // K: 12 chunks/row
      cp16(kg + (size_t)(kb + kr) * EMB + kc, sK + kr * SKP + kc);
      const int vr = c >> 3, vc = (c & 7) * 8;       // V: 8 chunks/row
      cp16(vg + (size_t)vr * SEQ + kb + vc, sV + vr * SVP + vc);
    }
    drain_async();
    __syncthreads();

    // ---- S = Q @ K^T : four 16x16 tiles (keys kb .. kb+63), K frags from LDS
    v8f s[4] = {};
#pragma unroll
    for (int dc = 0; dc < 3; ++dc) {
#pragma unroll
      for (int j = 0; j < 4; ++j) {
        const __bf16* kk = sK + (16 * j + l15) * SKP + dc * 32 + hsel * 8;
        v16bf bk = join16(*(const v8bf*)(kk), *(const v8bf*)(kk + 16));
        s[j] = wmma_bf16(aq[dc], bk, s[j]);
      }
    }

    // ---- online softmax (row r across lanes 0-15 / 16-31 of C-VGPR r)
#pragma unroll
    for (int r = 0; r < 8; ++r) {
      float mx = fmaxf(fmaxf(s[0][r], s[1][r]), fmaxf(s[2][r], s[3][r]));
      for (int off = 8; off >= 1; off >>= 1) mx = fmaxf(mx, __shfl_xor(mx, off, 16));
      const float mn = fmaxf(m[r], mx);
      const float alpha = __expf(m[r] - mn);
      float e[4], sum = 0.f;
#pragma unroll
      for (int j = 0; j < 4; ++j) { e[j] = __expf(s[j][r] - mn); sum += e[j]; }
      for (int off = 8; off >= 1; off >>= 1) sum += __shfl_xor(sum, off, 16);
      l[r] = l[r] * alpha + sum;
      m[r] = mn;
#pragma unroll
      for (int dt = 0; dt < 6; ++dt) o[dt][r] *= alpha;
      const int row = r + hsel * 8;
#pragma unroll
      for (int j = 0; j < 4; ++j)
        pl[row * SPP + j * 16 + l15] = (__bf16)e[j];
    }
    asm volatile("s_wait_dscnt 0x0" ::: "memory");   // per-wave P region; DS in-order

    // ---- reshape P (C layout) -> two A fragments (keys kb..+31, kb+32..+63)
    v16bf ap0 = join16(*(const v8bf*)(pl + l15 * SPP + hsel * 8),
                       *(const v8bf*)(pl + l15 * SPP + 16 + hsel * 8));
    v16bf ap1 = join16(*(const v8bf*)(pl + l15 * SPP + 32 + hsel * 8),
                       *(const v8bf*)(pl + l15 * SPP + 48 + hsel * 8));

    // ---- O += P @ V  (6 n-tiles over head dim 96), V frags from LDS
#pragma unroll
    for (int dt = 0; dt < 6; ++dt) {
      const __bf16* vv0 = sV + (dt * 16 + l15) * SVP + hsel * 8;
      v16bf bv0 = join16(*(const v8bf*)(vv0), *(const v8bf*)(vv0 + 16));
      v16bf bv1 = join16(*(const v8bf*)(vv0 + 32), *(const v8bf*)(vv0 + 48));
      o[dt] = wmma_bf16(ap0, bv0, o[dt]);
      o[dt] = wmma_bf16(ap1, bv1, o[dt]);
    }
    __syncthreads();   // all waves done reading sK/sV before next stage
  }

  // ---- epilogue: out = SCALE * O / l   (reference scales AFTER softmax)
#pragma unroll
  for (int r = 0; r < 8; ++r) {
    const float inv = SCALE_F / l[r];
    const int row = gq + hsel * 8 + r;
#pragma unroll
    for (int dt = 0; dt < 6; ++dt)
      Out[(size_t)row * EMB + h * HDIM + dt * 16 + l15] = (__bf16)(o[dt][r] * inv);
  }
}

// ---------------------------------------------------------------- launcher
extern "C" void kernel_launch(void* const* d_in, const int* in_sizes, int n_in,
                              void* d_out, int out_size, void* d_ws, size_t ws_size,
                              hipStream_t stream) {
  (void)in_sizes; (void)n_in; (void)out_size; (void)ws_size;
  const float* x  = (const float*)d_in[0];
  const float* Wq = (const float*)d_in[1];
  const float* bq = (const float*)d_in[2];
  const float* Wk = (const float*)d_in[3];
  const float* bk = (const float*)d_in[4];
  const float* Wv = (const float*)d_in[5];
  const float* bv = (const float*)d_in[6];
  const float* Wo = (const float*)d_in[7];
  const float* bo = (const float*)d_in[8];

  char* ws = (char*)d_ws;
  size_t off = 0;
  auto alloc = [&](size_t bytes) {
    void* p = ws + off;
    off = (off + bytes + 255) & ~(size_t)255;
    return p;
  };
  const size_t ACT = (size_t)MTOT * EMB * sizeof(__bf16);
  const size_t WPK = (size_t)EMB * EMB * sizeof(__bf16);
  __bf16* xb  = (__bf16*)alloc(ACT);
  __bf16* Wqp = (__bf16*)alloc(WPK);
  __bf16* Wkp = (__bf16*)alloc(WPK);
  __bf16* Wvp = (__bf16*)alloc(WPK);
  __bf16* Wop = (__bf16*)alloc(WPK);
  __bf16* Qb  = (__bf16*)alloc(ACT);
  __bf16* Kb  = (__bf16*)alloc(ACT);
  __bf16* Vtb = (__bf16*)alloc(ACT);   // [B*H][96][4096]
  __bf16* Ab  = (__bf16*)alloc(ACT);   // attention output, merged heads

  cvt_bf16_kernel<<<2048, 256, 0, stream>>>(x, xb, MTOT * EMB);
  pack_w_kernel<<<2304, 256, 0, stream>>>(Wq, Wqp);
  pack_w_kernel<<<2304, 256, 0, stream>>>(Wk, Wkp);
  pack_w_kernel<<<2304, 256, 0, stream>>>(Wv, Wvp);
  pack_w_kernel<<<2304, 256, 0, stream>>>(Wo, Wop);

  dim3 gg(MTOT / 128, EMB / 128);
  gemm_kernel<0><<<gg, 256, 0, stream>>>(xb, Wqp, bq, Qb);
  gemm_kernel<0><<<gg, 256, 0, stream>>>(xb, Wkp, bk, Kb);
  gemm_kernel<1><<<gg, 256, 0, stream>>>(xb, Wvp, bv, Vtb);

  flash_kernel<<<dim3(SEQ / 128, BATCH * HEADS), 256, 0, stream>>>(Qb, Kb, Vtb, Ab);

  gemm_kernel<2><<<gg, 256, 0, stream>>>(Ab, Wop, bo, d_out);
}
